// Att_3_layer1_beifen_11879879544275
// MI455X (gfx1250) — compile-verified, matrix operand order checked by reading, not run on hardware
//
#include <hip/hip_runtime.h>
#include <hip/hip_bf16.h>

typedef __attribute__((ext_vector_type(16))) _Float16 v16h;
typedef __attribute__((ext_vector_type(8)))  _Float16 v8h;
typedef __attribute__((ext_vector_type(8)))  float    v8f;
typedef unsigned int u32x4 __attribute__((ext_vector_type(4)));
typedef int          i32x8 __attribute__((ext_vector_type(8)));
typedef int          i32x4 __attribute__((ext_vector_type(4)));

// ---------------------------------------------------------------------------
// Problem constants: B=8, K=4, T=64 -> KT=256, O=100 (pad 112), VD=2048,
// QD=768, H=512.
// ---------------------------------------------------------------------------
#define KT      256
#define NB      8
#define O_REAL  100
#define O_PAD   112
#define HID     512
// LDS row stride for the staged vp tile: 512 halves + 8 halves (16B) pad so
// successive rows land 260 dwords apart -> conflict-free ds_load_b128.
#define VROW    520

#if __has_builtin(__builtin_amdgcn_tensor_load_to_lds) && \
    __has_builtin(__builtin_amdgcn_s_wait_tensorcnt)
#define USE_TDM 1
#else
#define USE_TDM 0
#endif

__device__ __forceinline__ v8f vzero8() {
  v8f z;
#pragma unroll
  for (int i = 0; i < 8; ++i) z[i] = 0.0f;
  return z;
}

// Load a 16-half WMMA fragment: elements 0..7 at p[0..7], elements 8..15 at
// p[16..23] (K-split layout for 16-bit A/B operands on CDNA5 wave32).
__device__ __forceinline__ v16h load_frag16(const _Float16* __restrict__ p) {
  v8h lo = *(const v8h*)p;
  v8h hi = *(const v8h*)(p + 16);
  return __builtin_shufflevector(lo, hi, 0,1,2,3,4,5,6,7,8,9,10,11,12,13,14,15);
}

// ---------------------------------------------------------------------------
// Utility kernels (bandwidth-trivial, run once)
// ---------------------------------------------------------------------------
__global__ __launch_bounds__(256) void zero_u32_kernel(unsigned* __restrict__ p, int n) {
  int i = blockIdx.x * 256 + threadIdx.x;
  if (i < n) p[i] = 0u;
}

__global__ __launch_bounds__(256) void cvt_f32_to_f16_kernel(const float* __restrict__ src,
                                                             _Float16* __restrict__ dst, int n) {
  int i = blockIdx.x * 256 + threadIdx.x;
  if (i < n) dst[i] = (_Float16)src[i];
}

// ---------------------------------------------------------------------------
// Projection GEMM: out = relu(X16 @ W16^T + bias), X16:[M,Kd] f16,
// W16:[HID,Kd] f16, out:[*,HID] f16. Each wave owns a 16x64 output strip
// (1 M-tile x 4 N-tiles): one A fragment feeds 4 v_wmma per k-step.
// If pad!=0, output rows are remapped 100 -> 112 per batch.
// ---------------------------------------------------------------------------
__global__ __launch_bounds__(256) void proj_gemm_relu_f16_kernel(
    const _Float16* __restrict__ X16, const _Float16* __restrict__ W16,
    const float* __restrict__ bias, _Float16* __restrict__ out,
    int M, int Kd, int pad) {
  const int wave = threadIdx.x >> 5;
  const int lane = threadIdx.x & 31;
  const int gw   = blockIdx.x * 8 + wave;
  const int tm   = gw >> 3;        // M-tile index (HID/64 = 8 N-groups)
  const int ng   = gw & 7;         // N-group: 64 outputs = 4 WMMA tiles
  if (tm * 16 >= M) return;        // whole wave exits together

  const int l15   = lane & 15;
  const int half  = lane >> 4;
  const int koff  = half * 8;
  const int nbase = ng * 64;

  const _Float16* arow = X16 + (size_t)(tm * 16 + l15) * Kd;

  v8f acc[4];
#pragma unroll
  for (int nt = 0; nt < 4; ++nt) acc[nt] = vzero8();

  for (int k0 = 0; k0 < Kd; k0 += 32) {
    v16h a = load_frag16(arow + k0 + koff);
#pragma unroll
    for (int nt = 0; nt < 4; ++nt) {
      v16h bf = load_frag16(W16 + (size_t)(nbase + nt * 16 + l15) * Kd + k0 + koff);
      acc[nt] = __builtin_amdgcn_wmma_f32_16x16x32_f16(false, a, false, bf,
                                                       (short)0, acc[nt], false, false);
    }
  }

#pragma unroll
  for (int nt = 0; nt < 4; ++nt) {
    const int   n  = nbase + nt * 16 + l15;
    const float bb = bias[n];
#pragma unroll
    for (int r = 0; r < 8; ++r) {
      int m    = tm * 16 + r + 8 * half;   // C layout: VGPR r -> row r (+8 hi half)
      int orow = pad ? ((m / O_REAL) * O_PAD + (m % O_REAL)) : m;
      float v  = acc[nt][r] + bb;
      v = v > 0.0f ? v : 0.0f;
      out[(size_t)orow * HID + n] = (_Float16)v;
    }
  }
}

// ---------------------------------------------------------------------------
// Fused main kernel: block = (kt, b). The vp[b] tile (112x512 f16) is staged
// into LDS once per block via the Tensor Data Mover (with pad_enable for a
// conflict-free 1040B row stride), then all 8 waves consume it as ds_load_b128
// fragments. Register blocking: 7 o-groups x 4 n-tiles accumulators per wave.
//   joint_frag = vtile_frag * qp_frag (broadcast)   [v_pk_mul_f16]
//   acc += v_wmma_f32_16x16x32_f16(joint_frag, Wn_frag)
// Epilogue: relu(acc+bn)*Wl -> shfl_xor butterfly -> ds_add_f32 -> softmax.
// ---------------------------------------------------------------------------
__global__ __launch_bounds__(256) void att_fused_kernel(
    const _Float16* __restrict__ vp,     // [NB, O_PAD, HID] f16 (padded rows 0)
    const _Float16* __restrict__ qp,     // [NB*KT, HID] f16
    const _Float16* __restrict__ wn16,   // [HID, HID] f16 (row g, col h)
    const float* __restrict__ bn, const float* __restrict__ Wl,
    const float* __restrict__ bl, const int* __restrict__ box_mask,
    float* __restrict__ outw) {
  const int kt   = blockIdx.x;           // 0..255 (= k*64 + t)
  const int b    = blockIdx.y;           // 0..7
  const int wave = threadIdx.x >> 5;
  const int lane = threadIdx.x & 31;
  const int l15  = lane & 15;
  const int half = lane >> 4;
  const int koff = half * 8;
  const int nwv  = wave * 64;            // this wave's N strip base

  __shared__ _Float16 vtile[O_PAD * VROW];   // 116,480 B staged vp[b] tile
  __shared__ float    sm[O_PAD];

  for (int t = threadIdx.x; t < O_PAD; t += 256) sm[t] = bl[0];

#if USE_TDM
  if (wave == 0) {
    // TDM D#: 1D copy of O_PAD*HID f16 = 14336 x 8B elements, LDS padding of
    // 4 dwords every 256 dwords (pad_interval=7, pad_amount=3) -> VROW stride.
    const unsigned long long ga =
        (unsigned long long)(uintptr_t)(vp + (size_t)b * O_PAD * HID);
    const unsigned lds = (unsigned)(uintptr_t)&vtile[0];
    const int n8 = O_PAD * HID * 2 / 8;  // 14336 8-byte elements

    u32x4 g0;
    g0[0] = 1u;                                   // count=1, user descriptor
    g0[1] = lds;                                  // lds_addr (bytes)
    g0[2] = (unsigned)(ga & 0xFFFFFFFFu);         // global_addr[31:0]
    g0[3] = (unsigned)((ga >> 32) & 0x01FFFFFFu)  // global_addr[56:32]
            | 0x80000000u;                        // type=2 ("image")

    i32x8 g1;
    g1[0] = (3 << 16)      // data_size = 3 (8B)
          | (1 << 20)      // pad_enable
          | (7 << 22)      // pad_interval: every 256 dwords
          | (3 << 25);     // pad_amount: 4 dwords
    g1[1] = (n8 & 0xFFFF) << 16;          // tensor_dim0[15:0] (bits 63:48)
    g1[2] = (n8 >> 16) | (1 << 16);       // tensor_dim0[31:16] | tensor_dim1=1
    g1[3] = (n8 & 0xFFFF) << 16;          // tile_dim0 (bits 127:112)
    g1[4] = 1;                            // tile_dim1 = 1, tile_dim2 = 0
    g1[5] = n8;                           // tensor_dim0_stride[31:0]
    g1[6] = (n8 & 0xFFFF) << 16;          // dim0_stride[47:32]=0 | dim1_stride lo
    g1[7] = 0;

    i32x4 gz4;
    gz4[0] = 0; gz4[1] = 0; gz4[2] = 0; gz4[3] = 0;
    i32x8 gz8;
#pragma unroll
    for (int i = 0; i < 8; ++i) gz8[i] = 0;

    __builtin_amdgcn_tensor_load_to_lds(g0, g1, gz4, gz4, gz8, 0);
    __builtin_amdgcn_s_wait_tensorcnt(0);
  }
#else
  {
    const unsigned long long* src =
        (const unsigned long long*)(vp + (size_t)b * O_PAD * HID);
    for (int i = threadIdx.x; i < O_PAD * HID / 4; i += 256) {
      int row = i >> 7, within = i & 127;
      *(unsigned long long*)&vtile[row * VROW + within * 4] = src[i];
    }
  }
#endif
  __syncthreads();

  v8f acc[7][4];
#pragma unroll
  for (int og = 0; og < 7; ++og)
#pragma unroll
    for (int nt = 0; nt < 4; ++nt) acc[og][nt] = vzero8();

  const _Float16* qrow = qp + ((size_t)(b * KT + kt)) * HID;

  for (int k0 = 0; k0 < HID; k0 += 32) {
    v16h qf = load_frag16(qrow + k0 + koff);
    v16h bf[4];
#pragma unroll
    for (int nt = 0; nt < 4; ++nt)
      bf[nt] = load_frag16(wn16 + (size_t)(nwv + nt * 16 + l15) * HID + k0 + koff);
#pragma unroll
    for (int og = 0; og < 7; ++og) {
      v16h vf = load_frag16(&vtile[(og * 16 + l15) * VROW + k0 + koff]);  // ds_load_b128
      v16h a  = vf * qf;   // joint fragment, never materialized in memory
#pragma unroll
      for (int nt = 0; nt < 4; ++nt)
        acc[og][nt] = __builtin_amdgcn_wmma_f32_16x16x32_f16(
            false, a, false, bf[nt], (short)0, acc[og][nt], false, false);
    }
  }

  // Epilogue: relu(acc + bn) * Wl; butterfly-reduce across the 16 lanes that
  // share the same rows (xor masks 1..8 stay within each 16-lane half), then
  // 2 lanes per wave commit with ds_add_f32.
#pragma unroll
  for (int og = 0; og < 7; ++og) {
    float p[8];
#pragma unroll
    for (int r = 0; r < 8; ++r) p[r] = 0.0f;
#pragma unroll
    for (int nt = 0; nt < 4; ++nt) {
      int   g   = nwv + nt * 16 + l15;
      float bng = bn[g];
      float wlg = Wl[g];
#pragma unroll
      for (int r = 0; r < 8; ++r) {
        float v = acc[og][nt][r] + bng;
        v = v > 0.0f ? v : 0.0f;
        p[r] += v * wlg;
      }
    }
#pragma unroll
    for (int mph = 1; mph <= 8; mph <<= 1)
#pragma unroll
      for (int r = 0; r < 8; ++r) p[r] += __shfl_xor(p[r], mph, 32);
    if (l15 == 0) {
#pragma unroll
      for (int r = 0; r < 8; ++r)
        atomicAdd(&sm[og * 16 + r + 8 * half], p[r]);   // ds_add_f32
    }
  }
  __syncthreads();

  // Masked softmax over O (wave 0). Padding rows -> -inf (weight 0); masked
  // boxes -> -1e9 (reference NEG fill; all-masked -> uniform, matching ref).
  if (wave == 0) {
    float mx = -INFINITY;
    for (int o = lane; o < O_PAD; o += 32) {
      float lg = (o < O_REAL)
                     ? ((box_mask[b * O_REAL + o] > 0) ? sm[o] : -1e9f)
                     : -INFINITY;
      sm[o] = lg;
      mx = fmaxf(mx, lg);
    }
#pragma unroll
    for (int i = 16; i > 0; i >>= 1) mx = fmaxf(mx, __shfl_xor(mx, i, 32));
    float s = 0.0f;
    for (int o = lane; o < O_PAD; o += 32) {
      float e = __expf(sm[o] - mx);
      sm[o] = e;
      s += e;
    }
#pragma unroll
    for (int i = 16; i > 0; i >>= 1) s += __shfl_xor(s, i, 32);
    float inv = 1.0f / s;
    for (int o = lane; o < O_REAL; o += 32)
      outw[((size_t)(b * KT + kt)) * O_REAL + o] = sm[o] * inv;
  }
}

// ---------------------------------------------------------------------------
// Launcher
// ---------------------------------------------------------------------------
extern "C" void kernel_launch(void* const* d_in, const int* in_sizes, int n_in,
                              void* d_out, int out_size, void* d_ws, size_t ws_size,
                              hipStream_t stream) {
  (void)in_sizes; (void)n_in; (void)out_size; (void)ws_size;

  const float* v  = (const float*)d_in[0];   // [8,100,2048]
  const float* q  = (const float*)d_in[1];   // [8,4,64,768]
  const int*   bm = (const int*)d_in[2];     // [8,100]
  const float* Wv = (const float*)d_in[3];   // [512,2048]
  const float* bv = (const float*)d_in[4];   // [512]
  const float* Wq = (const float*)d_in[5];   // [512,768]
  const float* bq = (const float*)d_in[6];   // [512]
  const float* Wn = (const float*)d_in[7];   // [512,512]
  const float* bn = (const float*)d_in[8];   // [512]
  const float* Wl = (const float*)d_in[9];   // [1,512]
  const float* bl = (const float*)d_in[10];  // [1]

  // Workspace layout (all f16, every region a multiple of 256 B):
  //   vp   @ 0        : [8][112][512]   =   917,504 B
  //   qp   @ 917504   : [2048][512]     = 2,097,152 B
  //   wn16 @ 3014656  : [512][512]      =   524,288 B
  //   v16  @ 3538944  : [800][2048]     = 3,276,800 B
  //   q16  @ 6815744  : [2048][768]     = 3,145,728 B
  //   wv16 @ 9961472  : [512][2048]     = 2,097,152 B
  //   wq16 @ 12058624 : [512][768]      =   786,432 B   (end: 12,845,056 B)
  char* ws = (char*)d_ws;
  _Float16* vp   = (_Float16*)(ws);
  _Float16* qp   = (_Float16*)(ws + 917504);
  _Float16* wn16 = (_Float16*)(ws + 3014656);
  _Float16* v16  = (_Float16*)(ws + 3538944);
  _Float16* q16  = (_Float16*)(ws + 6815744);
  _Float16* wv16 = (_Float16*)(ws + 9961472);
  _Float16* wq16 = (_Float16*)(ws + 12058624);

  // Zero vp (padded o-rows 100..111 must be 0 for the fused GEMM).
  {
    int n_u32 = (NB * O_PAD * HID * 2) / 4;  // 229376
    zero_u32_kernel<<<(n_u32 + 255) / 256, 256, 0, stream>>>((unsigned*)vp, n_u32);
  }

  // One-shot f32 -> f16 conversions (keeps converts out of the GEMM loops).
  cvt_f32_to_f16_kernel<<<(800 * 2048 + 255) / 256, 256, 0, stream>>>(v,  v16,  800 * 2048);
  cvt_f32_to_f16_kernel<<<(2048 * 768 + 255) / 256, 256, 0, stream>>>(q,  q16,  2048 * 768);
  cvt_f32_to_f16_kernel<<<(512 * 2048 + 255) / 256, 256, 0, stream>>>(Wv, wv16, 512 * 2048);
  cvt_f32_to_f16_kernel<<<(512 * 768 + 255) / 256, 256, 0, stream>>>(Wq, wq16, 512 * 768);
  cvt_f32_to_f16_kernel<<<(512 * 512 + 255) / 256, 256, 0, stream>>>(Wn, wn16, 512 * 512);

  // v_proj: M=800 -> 50 M-tiles x 8 N-groups = 400 waves -> 50 blocks.
  proj_gemm_relu_f16_kernel<<<50, 256, 0, stream>>>(v16, wv16, bv, vp, 800, 2048, 1);
  // q_proj: M=2048 -> 128 M-tiles x 8 N-groups = 1024 waves -> 128 blocks.
  proj_gemm_relu_f16_kernel<<<128, 256, 0, stream>>>(q16, wq16, bq, qp, 2048, 768, 0);

  // Fused joint * Wn -> relu -> dot(Wl) -> masked softmax.
  att_fused_kernel<<<dim3(KT, NB), 256, 0, stream>>>(vp, qp, wn16, bn, Wl, bl, bm,
                                                     (float*)d_out);
}
